// TransGenerator_71141838291434
// MI455X (gfx1250) — matrix operand order repaired and checked by reading, not compiled, not run on hardware
//
#include <hip/hip_runtime.h>

// ---------------------------------------------------------------------------
// TransGenerator for MI455X (gfx1250):
//   - all GEMM/attention operands kept as bf16 in global memory
//   - LDS staging via GLOBAL_LOAD_ASYNC_TO_LDS_B128 (ASYNCcnt), double-buffered
//     with a peeled, branch-free steady-state loop
//   - v_wmma_f32_16x16x32_bf16 for all matrix math, f32 accumulate
// ---------------------------------------------------------------------------
#define B_   2
#define S_   1024
#define E_   1024
#define H_   16
#define L_   6
#define F_   4096
#define V_   400
#define DH_  64
#define QKV3 (3 * E_)

typedef __attribute__((ext_vector_type(16))) __bf16        v16bf;
typedef __attribute__((ext_vector_type(8)))  float         v8f;
typedef __attribute__((ext_vector_type(4)))  float         f32x4;
typedef __attribute__((ext_vector_type(4)))  unsigned int  u32x4;
typedef __attribute__((ext_vector_type(2)))  unsigned int  u32x2;

union Frag { u32x4 u[2]; v16bf v; };   // 16 bf16 = one WMMA A/B fragment per lane

__device__ __forceinline__ unsigned short f2bf(float f) {
  union { float f; unsigned u; } a; a.f = f;
  unsigned r = a.u + 0x7FFFu + ((a.u >> 16) & 1u);  // round-to-nearest-even
  return (unsigned short)(r >> 16);
}
__device__ __forceinline__ unsigned pack2(float a, float b) {
  return (unsigned)f2bf(a) | ((unsigned)f2bf(b) << 16);
}

// async 16-byte global -> LDS copy (per lane), tracked by ASYNCcnt
#define ASYNC_LD16(ldsOff, gptr)                                          \
  asm volatile("global_load_async_to_lds_b128 %0, %1, off"                \
               :: "v"(ldsOff),                                            \
                  "v"((unsigned long long)(gptr))                         \
               : "memory")
#define WAIT_ASYNC_0() asm volatile("s_wait_asynccnt 0x0" ::: "memory")
#define WAIT_ASYNC_1() asm volatile("s_wait_asynccnt 0x1" ::: "memory")
#define WAIT_ASYNC_4() asm volatile("s_wait_asynccnt 0x4" ::: "memory")

__device__ __forceinline__ unsigned lds_off(const void* p) {
  return (unsigned)(unsigned long long)p;   // LDS aperture: low 32 bits = offset
}

// ---------------------------------------------------------------------------
// fp32 -> bf16 weight conversion (one-time; bf16 weights ~145MB < 192MB L2)
// ---------------------------------------------------------------------------
__global__ void cvt_bf16_kernel(const float* __restrict__ in,
                                unsigned short* __restrict__ out, size_t n4) {
  size_t i = (size_t)blockIdx.x * blockDim.x + threadIdx.x;
  size_t stride = (size_t)gridDim.x * blockDim.x;
  for (; i < n4; i += stride) {
    f32x4 v = ((const f32x4*)in)[i];
    u32x2 o; o.x = pack2(v.x, v.y); o.y = pack2(v.z, v.w);
    ((u32x2*)out)[i] = o;
  }
}

// ---------------------------------------------------------------------------
// Token embedding -> x (fp32, residual path) + x_bf (bf16, GEMM operand path)
// ---------------------------------------------------------------------------
__global__ void embed_kernel(const int* __restrict__ seqs, const int* __restrict__ map1,
                             const int* __restrict__ map2, const float* __restrict__ emb,
                             const float* __restrict__ pos, float* __restrict__ x,
                             unsigned short* __restrict__ x_bf) {
  int row = blockIdx.x;            // 0 .. B*S-1
  int s   = row & (S_ - 1);
  int tok = seqs[row];
  int i1 = map1[tok], i2 = map2[tok];
  const f32x4* e1 = (const f32x4*)(emb + (size_t)i1 * E_);
  const f32x4* e2 = (const f32x4*)(emb + (size_t)i2 * E_);
  const f32x4* pp = (const f32x4*)(pos + (size_t)s * E_);
  f32x4* xo = (f32x4*)(x + (size_t)row * E_);
  u32x2* xb = (u32x2*)(x_bf + (size_t)row * E_);
  for (int c = threadIdx.x; c < E_ / 4; c += blockDim.x) {
    f32x4 o = (e1[c] + e2[c]) * 16.0f + pp[c];   // sqrt(1024)/2 = 16
    xo[c] = o;
    u32x2 ob; ob.x = pack2(o.x, o.y); ob.y = pack2(o.z, o.w);
    xb[c] = ob;
  }
}

// ---------------------------------------------------------------------------
// WMMA GEMM: act(A[M,K]bf16 @ W[N,K]bf16^T + bias) -> C (fp32) and/or C2 (bf16)
// 128x128 block tile, 8 waves (4x2), wave tile 32x64, K-step 32.
// Double-buffered LDS fed by async global->LDS copies; last iteration peeled
// so the steady-state loop is branch-free.
// ---------------------------------------------------------------------------
__global__ __launch_bounds__(256)
void gemm_wmma_kernel(const unsigned short* __restrict__ A,
                      const unsigned short* __restrict__ W,
                      const float* __restrict__ bias,
                      float* __restrict__ C, unsigned short* __restrict__ C2,
                      int M, int N, int K, int act) {
  __shared__ unsigned short As[2][128 * 40];   // 32 K + 8 pad (80B rows, 16B aligned)
  __shared__ unsigned short Bs[2][128 * 40];
  const int tid  = threadIdx.x;
  const int lane = tid & 31;
  const int wave = tid >> 5;
  const int wm = wave >> 1, wn = wave & 1;
  const int half = lane >> 4, r = lane & 15;
  const int m0 = blockIdx.y * 128, n0 = blockIdx.x * 128;

  v8f acc[2][4];
  for (int i = 0; i < 2; i++) for (int j = 0; j < 4; j++) acc[i][j] = {};

  const int srow = tid >> 1;                 // staging: 128 rows x 2 half-rows
  const int scol = (tid & 1) * 16;
  int gr = m0 + srow; if (gr >= M) gr = M - 1;   // clamp: every lane always copies
  int gn = n0 + srow; if (gn >= N) gn = N - 1;   //  -> exactly 4 async ops per lane
  const unsigned short* gA = A + (size_t)gr * K + scol;
  const unsigned short* gW = W + (size_t)gn * K + scol;
  const unsigned laBase = lds_off(&As[0][0]) + (unsigned)(srow * 40 + scol) * 2;
  const unsigned lbBase = lds_off(&Bs[0][0]) + (unsigned)(srow * 40 + scol) * 2;
  const unsigned TILE_B = 128 * 40 * 2;

  auto stage = [&](int buf, int k0) {
    unsigned la = laBase + (unsigned)buf * TILE_B;
    unsigned lb = lbBase + (unsigned)buf * TILE_B;
    ASYNC_LD16(la,      gA + k0);
    ASYNC_LD16(la + 16, gA + k0 + 8);
    ASYNC_LD16(lb,      gW + k0);
    ASYNC_LD16(lb + 16, gW + k0 + 8);
  };
  auto compute = [&](int buf) {
    const unsigned short* Ab = As[buf];
    const unsigned short* Bb = Bs[buf];
    Frag af[2], bf[4];
    for (int mi = 0; mi < 2; mi++) {            // A 16x32: lane<16 K{0-7,16-23}
      int off = (wm * 32 + mi * 16 + r) * 40 + half * 8;
      af[mi].u[0] = *(const u32x4*)(&Ab[off]);
      af[mi].u[1] = *(const u32x4*)(&Ab[off + 16]);
    }
    for (int ni = 0; ni < 4; ni++) {            // B 32x16: lane<16 K{0-7,16-23}
      int off = (wn * 64 + ni * 16 + r) * 40 + half * 8;
      bf[ni].u[0] = *(const u32x4*)(&Bb[off]);
      bf[ni].u[1] = *(const u32x4*)(&Bb[off + 16]);
    }
    for (int mi = 0; mi < 2; mi++)
      for (int ni = 0; ni < 4; ni++)
        acc[mi][ni] = __builtin_amdgcn_wmma_f32_16x16x32_bf16(
            false, af[mi].v, false, bf[ni].v, (short)0, acc[mi][ni], false, false);
  };

  const int nk = K >> 5;     // always >= 32 here
  stage(0, 0);
  for (int i = 0; i < nk - 1; ++i) {   // branch-free steady state
    stage((i + 1) & 1, (i + 1) << 5);
    WAIT_ASYNC_4();
    __syncthreads();
    compute(i & 1);
    __syncthreads();
  }
  WAIT_ASYNC_0();                      // peeled tail
  __syncthreads();
  compute((nk - 1) & 1);

  // epilogue: D layout lane l -> col (l&15), rows rr + 8*(l>>4)
  for (int ni = 0; ni < 4; ni++) {
    int col = n0 + wn * 64 + ni * 16 + r;
    if (col >= N) continue;
    float bb = bias[col];
    for (int mi = 0; mi < 2; mi++) {
      int rbase = m0 + wm * 32 + mi * 16 + 8 * half;
      for (int rr = 0; rr < 8; rr++) {
        int row = rbase + rr;
        if (row < M) {
          float v = acc[mi][ni][rr] + bb;
          if (act) v = 0.5f * v * (1.0f + erff(v * 0.70710678118654752f));  // exact gelu
          size_t idx = (size_t)row * N + col;
          if (C)  C[idx]  = v;
          if (C2) C2[idx] = f2bf(v);
        }
      }
    }
  }
}

// ---------------------------------------------------------------------------
// Fused flash attention (causal + relative-distance bias + pad mask).
// Block = (b, h, 128 query rows). 8 waves x 16 rows. 32-key tiles.
// Double-buffered K (async copy) and V (transpose-staged during compute).
// ---------------------------------------------------------------------------
__global__ __launch_bounds__(256)
void attention_kernel(const unsigned short* __restrict__ qkv, const int* __restrict__ seqs,
                      const float* __restrict__ dist_emb,
                      unsigned short* __restrict__ ctx_bf) {
  __shared__ unsigned short Ks[2][32 * 80];    // [key][dim 64 + pad]
  __shared__ unsigned short Vs[2][64 * 40];    // V^T: [dim][key 32 + pad]
  __shared__ unsigned short Ps[8 * 16 * 40];   // per-wave P tile [16][32 + pad]
  const int tid = threadIdx.x, lane = tid & 31, wave = tid >> 5;
  const int half = lane >> 4, r = lane & 15;
  const int h = blockIdx.y, b = blockIdx.z;
  const int qBase = blockIdx.x * 128;

  // Q fragments: 2 x (16x32) per wave, straight bf16 loads (scale applied later)
  Frag qf[2];
  {
    int rowq = qBase + wave * 16 + r;
    const unsigned short* qp = qkv + (size_t)(b * S_ + rowq) * QKV3 + h * DH_;
    for (int kk = 0; kk < 2; kk++) {
      const unsigned short* base = qp + kk * 32 + half * 8;
      qf[kk].u[0] = *(const u32x4*)(base);
      qf[kk].u[1] = *(const u32x4*)(base + 16);
    }
  }

  float m[8], lsum[8];
  for (int i = 0; i < 8; i++) { m[i] = -__builtin_inff(); lsum[i] = 0.0f; }
  v8f acc[4];
  for (int g = 0; g < 4; g++) acc[g] = {};

  const int pb = wave * 16 * 40;    // this wave's private P region
  const unsigned ksBase = lds_off(&Ks[0][0]);
  const unsigned KTILE_B = 32 * 80 * 2;
  const int kkey = tid >> 3, kdimB = (tid & 7) * 8;
  const int vdim = tid >> 2, vkb = (tid & 3) * 8;

  auto stageK = [&](int buf, int j0) {   // one async b128 per lane
    ASYNC_LD16(ksBase + (unsigned)buf * KTILE_B + (unsigned)(kkey * 80 + kdimB) * 2,
               qkv + (size_t)(b * S_ + j0 + kkey) * QKV3 + E_ + h * DH_ + kdimB);
  };
  auto stageV = [&](int buf, int j0) {   // transpose during bf16 staging
    const unsigned short* vp =
        qkv + (size_t)(b * S_ + j0 + vkb) * QKV3 + 2 * E_ + h * DH_ + vdim;
    u32x4 sv;
    sv.x = (unsigned)vp[0]        | ((unsigned)vp[QKV3]     << 16);
    sv.y = (unsigned)vp[2 * QKV3] | ((unsigned)vp[3 * QKV3] << 16);
    sv.z = (unsigned)vp[4 * QKV3] | ((unsigned)vp[5 * QKV3] << 16);
    sv.w = (unsigned)vp[6 * QKV3] | ((unsigned)vp[7 * QKV3] << 16);
    *(u32x4*)(&Vs[buf][vdim * 40 + vkb]) = sv;
  };

  auto compute = [&](int buf, int j0) {
    // S = Q K^T (two 16x16 score tiles covering 32 keys)
    v8f st[2];
    for (int c = 0; c < 2; c++) {
      v8f s = {};
      for (int kk = 0; kk < 2; kk++) {
        Frag bfz;
        int off = (c * 16 + r) * 80 + kk * 32 + half * 8;
        bfz.u[0] = *(const u32x4*)(&Ks[buf][off]);
        bfz.u[1] = *(const u32x4*)(&Ks[buf][off + 16]);
        s = __builtin_amdgcn_wmma_f32_16x16x32_bf16(
            false, qf[kk].v, false, bfz.v, (short)0, s, false, false);
      }
      // 1/sqrt(DH) scale + bias + causal + pad mask
      int j = j0 + c * 16 + r;
      bool pad = (seqs[b * S_ + j] == 0);
      for (int rr = 0; rr < 8; rr++) {
        int i = qBase + wave * 16 + 8 * half + rr;
        int d = i - j;
        float v = s[rr] * 0.125f;
        if (d < 0 || pad) v = -__builtin_inff();
        else              v += dist_emb[d * H_ + h];
        s[rr] = v;
      }
      st[c] = s;
    }

    // online softmax (16-lane reductions inside each half-wave)
    float tmax[8], psum[8], scl[8];
    for (int rr = 0; rr < 8; rr++) tmax[rr] = fmaxf(st[0][rr], st[1][rr]);
    for (int off = 1; off < 16; off <<= 1)
      for (int rr = 0; rr < 8; rr++)
        tmax[rr] = fmaxf(tmax[rr], __shfl_xor(tmax[rr], off, 32));
    for (int rr = 0; rr < 8; rr++) {
      float mn = fmaxf(m[rr], tmax[rr]);
      scl[rr] = __expf(m[rr] - mn);
      m[rr] = mn;
      float p0 = __expf(st[0][rr] - mn);
      float p1 = __expf(st[1][rr] - mn);
      st[0][rr] = p0; st[1][rr] = p1;
      psum[rr] = p0 + p1;
    }
    for (int off = 1; off < 16; off <<= 1)
      for (int rr = 0; rr < 8; rr++)
        psum[rr] += __shfl_xor(psum[rr], off, 32);
    for (int rr = 0; rr < 8; rr++) lsum[rr] = lsum[rr] * scl[rr] + psum[rr];
    for (int g = 0; g < 4; g++)
      for (int rr = 0; rr < 8; rr++) acc[g][rr] *= scl[rr];

    // P -> LDS (wave-private region), then reload as A fragment
    for (int rr = 0; rr < 8; rr++) {
      int rowl = 8 * half + rr;
      Ps[pb + rowl * 40 + r]      = f2bf(st[0][rr]);
      Ps[pb + rowl * 40 + 16 + r] = f2bf(st[1][rr]);
    }
    asm volatile("s_wait_dscnt 0" ::: "memory");   // wave-local LDS RAW fence
    Frag pf;
    {
      int off = pb + r * 40 + half * 8;
      pf.u[0] = *(const u32x4*)(&Ps[off]);
      pf.u[1] = *(const u32x4*)(&Ps[off + 16]);
    }
    // acc += P V
    for (int g = 0; g < 4; g++) {
      Frag vf;
      int off = (g * 16 + r) * 40 + half * 8;
      vf.u[0] = *(const u32x4*)(&Vs[buf][off]);
      vf.u[1] = *(const u32x4*)(&Vs[buf][off + 16]);
      acc[g] = __builtin_amdgcn_wmma_f32_16x16x32_bf16(
          false, pf.v, false, vf.v, (short)0, acc[g], false, false);
    }
  };

  const int nt = (qBase + 128) >> 5;   // causal tile count, always >= 4
  stageK(0, 0);
  stageV(0, 0);
  for (int t = 0; t < nt - 1; ++t) {   // branch-free steady state
    int j1 = (t + 1) << 5;
    stageK((t + 1) & 1, j1);           // async prefetch next K tile
    WAIT_ASYNC_1();                    // current K tile resident
    __syncthreads();
    stageV((t + 1) & 1, j1);           // V transpose overlaps compute
    compute(t & 1, t << 5);
    __syncthreads();
  }
  WAIT_ASYNC_0();                      // peeled tail
  __syncthreads();
  compute((nt - 1) & 1, (nt - 1) << 5);

  // normalize and scatter ctx_bf[b, i, h*64 + dim]
  int rowq = qBase + wave * 16 + 8 * half;
  for (int g = 0; g < 4; g++) {
    int dim = g * 16 + r;
    for (int rr = 0; rr < 8; rr++) {
      int i = rowq + rr;
      ctx_bf[(size_t)(b * S_ + i) * E_ + h * DH_ + dim] = f2bf(acc[g][rr] / lsum[rr]);
    }
  }
}

// ---------------------------------------------------------------------------
// LayerNorm(x [+ delta]) -> out (fp32 residual path) + out_bf (bf16 GEMM path)
// ---------------------------------------------------------------------------
__global__ __launch_bounds__(256)
void add_ln_kernel(const float* __restrict__ x, const float* __restrict__ delta,
                   const float* __restrict__ w, const float* __restrict__ bgain,
                   float* __restrict__ out, unsigned short* __restrict__ out_bf,
                   int hasDelta) {
  __shared__ float red[16];
  int row = blockIdx.x, tid = threadIdx.x;
  f32x4 v = ((const f32x4*)(x + (size_t)row * E_))[tid];
  if (hasDelta) v += ((const f32x4*)(delta + (size_t)row * E_))[tid];
  float s  = v.x + v.y + v.z + v.w;
  float sq = v.x * v.x + v.y * v.y + v.z * v.z + v.w * v.w;
  for (int off = 1; off < 32; off <<= 1) {
    s  += __shfl_xor(s, off, 32);
    sq += __shfl_xor(sq, off, 32);
  }
  int wave = tid >> 5, lane = tid & 31;
  if (lane == 0) { red[wave] = s; red[8 + wave] = sq; }
  __syncthreads();
  if (tid < 32) {
    float a  = (tid < 8) ? red[tid] : 0.0f;
    float b2 = (tid < 8) ? red[8 + tid] : 0.0f;
    for (int off = 1; off < 8; off <<= 1) {
      a += __shfl_xor(a, off, 32);
      b2 += __shfl_xor(b2, off, 32);
    }
    if (tid == 0) { red[0] = a; red[1] = b2; }
  }
  __syncthreads();
  float mean = red[0] * (1.0f / E_);
  float var  = red[1] * (1.0f / E_) - mean * mean;
  float rs   = rsqrtf(var + 1e-5f);
  f32x4 wv = ((const f32x4*)w)[tid];
  f32x4 bv = ((const f32x4*)bgain)[tid];
  f32x4 o = (v - mean) * rs * wv + bv;
  ((f32x4*)(out + (size_t)row * E_))[tid] = o;
  u32x2 ob; ob.x = pack2(o.x, o.y); ob.y = pack2(o.z, o.w);
  ((u32x2*)(out_bf + (size_t)row * E_))[tid] = ob;
}

// ---------------------------------------------------------------------------
extern "C" void kernel_launch(void* const* d_in, const int* in_sizes, int n_in,
                              void* d_out, int out_size, void* d_ws, size_t ws_size,
                              hipStream_t stream) {
  (void)in_sizes; (void)n_in; (void)out_size; (void)ws_size;
  const int*   seqs = (const int*)d_in[0];
  const int*   map1 = (const int*)d_in[1];
  const int*   map2 = (const int*)d_in[2];
  const float* emb  = (const float*)d_in[3];
  const float* pos  = (const float*)d_in[4];
  const float* dist = (const float*)d_in[5];
  const float* in_w  = (const float*)d_in[6];
  const float* in_b  = (const float*)d_in[7];
  const float* out_w = (const float*)d_in[8];
  const float* out_b = (const float*)d_in[9];
  const float* l1w = (const float*)d_in[10];
  const float* l1b = (const float*)d_in[11];
  const float* l2w = (const float*)d_in[12];
  const float* l2b = (const float*)d_in[13];
  const float* ln1w = (const float*)d_in[14];
  const float* ln1b = (const float*)d_in[15];
  const float* ln2w = (const float*)d_in[16];
  const float* ln2b = (const float*)d_in[17];
  const float* fnw  = (const float*)d_in[18];
  const float* fnb  = (const float*)d_in[19];
  const float* genw = (const float*)d_in[20];
  const float* genb = (const float*)d_in[21];

  char* p = (char*)d_ws;
  auto alloc = [&](size_t bytes) -> char* {
    char* q = p; p += (bytes + 255) & ~(size_t)255; return q;
  };
  const size_t MB_ = (size_t)B_ * S_;   // 2048 rows
  unsigned short* w_in  = (unsigned short*)alloc((size_t)L_ * 3 * E_ * E_ * 2);
  unsigned short* w_out = (unsigned short*)alloc((size_t)L_ * E_ * E_ * 2);
  unsigned short* w_l1  = (unsigned short*)alloc((size_t)L_ * F_ * E_ * 2);
  unsigned short* w_l2  = (unsigned short*)alloc((size_t)L_ * E_ * F_ * 2);
  unsigned short* w_gen = (unsigned short*)alloc((size_t)V_ * E_ * 2);
  float* x   = (float*)alloc(MB_ * E_ * 4);
  float* x1  = (float*)alloc(MB_ * E_ * 4);
  float* tmp = (float*)alloc(MB_ * E_ * 4);
  unsigned short* x_bf   = (unsigned short*)alloc(MB_ * E_ * 2);
  unsigned short* x1_bf  = (unsigned short*)alloc(MB_ * E_ * 2);
  unsigned short* qkv_bf = (unsigned short*)alloc(MB_ * 3 * E_ * 2);
  unsigned short* ctx_bf = (unsigned short*)alloc(MB_ * E_ * 2);
  unsigned short* h_bf   = (unsigned short*)alloc(MB_ * F_ * 2);

  // one-time bf16 weight conversion (deterministic; redone every call)
  cvt_bf16_kernel<<<2048, 256, 0, stream>>>(in_w,  w_in,  (size_t)L_ * 3 * E_ * E_ / 4);
  cvt_bf16_kernel<<<2048, 256, 0, stream>>>(out_w, w_out, (size_t)L_ * E_ * E_ / 4);
  cvt_bf16_kernel<<<2048, 256, 0, stream>>>(l1w,   w_l1,  (size_t)L_ * F_ * E_ / 4);
  cvt_bf16_kernel<<<2048, 256, 0, stream>>>(l2w,   w_l2,  (size_t)L_ * E_ * F_ / 4);
  cvt_bf16_kernel<<<1024, 256, 0, stream>>>(genw,  w_gen, (size_t)V_ * E_ / 4);

  embed_kernel<<<(int)MB_, 256, 0, stream>>>(seqs, map1, map2, emb, pos, x, x_bf);

  dim3 blk(256);
  for (int i = 0; i < L_; i++) {
    gemm_wmma_kernel<<<dim3(24, 16), blk, 0, stream>>>(
        x_bf, w_in + (size_t)i * 3 * E_ * E_, in_b + (size_t)i * 3 * E_,
        nullptr, qkv_bf, 2048, 3 * E_, E_, 0);
    attention_kernel<<<dim3(8, H_, B_), blk, 0, stream>>>(qkv_bf, seqs, dist, ctx_bf);
    gemm_wmma_kernel<<<dim3(8, 16), blk, 0, stream>>>(
        ctx_bf, w_out + (size_t)i * E_ * E_, out_b + (size_t)i * E_,
        tmp, nullptr, 2048, E_, E_, 0);
    add_ln_kernel<<<(int)MB_, 256, 0, stream>>>(
        x, tmp, ln1w + i * E_, ln1b + i * E_, x1, x1_bf, 1);
    gemm_wmma_kernel<<<dim3(32, 16), blk, 0, stream>>>(
        x1_bf, w_l1 + (size_t)i * F_ * E_, l1b + (size_t)i * F_,
        nullptr, h_bf, 2048, F_, E_, 1);
    gemm_wmma_kernel<<<dim3(8, 16), blk, 0, stream>>>(
        h_bf, w_l2 + (size_t)i * E_ * F_, l2b + (size_t)i * E_,
        tmp, nullptr, 2048, E_, F_, 0);
    add_ln_kernel<<<(int)MB_, 256, 0, stream>>>(
        x1, tmp, ln2w + i * E_, ln2b + i * E_, x, x_bf, 1);
  }
  add_ln_kernel<<<(int)MB_, 256, 0, stream>>>(x, x, fnw, fnb, tmp, x1_bf, 0);
  gemm_wmma_kernel<<<dim3(4, 16), blk, 0, stream>>>(
      x1_bf, w_gen, genb, (float*)d_out, nullptr, 2048, V_, E_, 0);
}
